// PrecNetLearnAll_21028159881520
// MI455X (gfx1250) — matrix operand order repaired and checked by reading, not compiled, not run on hardware
//
#include <hip/hip_runtime.h>

#define HID 16
#define ROUNDS 5
#define NOPS (5 * ROUNDS + 1) // 26 prepared A-layout weight operands

typedef unsigned short u16;
typedef __attribute__((ext_vector_type(16))) __bf16 v16bf;
typedef __attribute__((ext_vector_type(16))) unsigned short v16u;
typedef __attribute__((ext_vector_type(8)))  unsigned short v8u;
typedef __attribute__((ext_vector_type(8)))  float v8f;

__device__ __forceinline__ u16 f2bf(float f) {
  unsigned int u = __float_as_uint(f);
  u += 0x7FFFu + ((u >> 16) & 1u); // round-to-nearest-even
  return (u16)(u >> 16);
}
__device__ __forceinline__ float bf2f(u16 h) {
  return __uint_as_float(((unsigned int)h) << 16);
}
// exchange with lane^16 (ds_swizzle SWAPX16: xor=0x10, and=0x1f)
__device__ __forceinline__ float swapx16(float x) {
  return __uint_as_float(
      (unsigned int)__builtin_amdgcn_ds_swizzle((int)__float_as_uint(x), 0x401F));
}
__device__ __forceinline__ v8f wmma_bf16(v16u a, v16u b, v8f c) {
  return __builtin_amdgcn_wmma_f32_16x16x32_bf16(
      false, __builtin_bit_cast(v16bf, a),
      false, __builtin_bit_cast(v16bf, b),
      (short)0, c, false, false);
}

// ---------------- weight repack into WMMA A-matrix layout ----------------
// A as packed here (16-bit, 16x32, contiguous k, not AMD's table order): lane L row M=L%16; element e -> K = e + 16*(L>=16)
// op 0..9   : mp_edge_W1[r] (16x48) -> 2 chunks (koff 0 / 32)
// op 10..14 : mp_edge_W2[r] (16x16)
// op 15..19 : mp_node_W1[r] (16x32)
// op 20..24 : mp_node_W2[r] (16x16)
// op 25     : edge_dec_W1   (16x16)
__global__ void prep_weights(const float* __restrict__ eW1, const float* __restrict__ eW2,
                             const float* __restrict__ nW1, const float* __restrict__ nW2,
                             const float* __restrict__ dW1, u16* __restrict__ wsA) {
  int op = blockIdx.x;
  int lane = threadIdx.x;
  const float* src;
  int Kw, koff = 0;
  if (op < 2 * ROUNDS)      { src = eW1 + (op >> 1) * HID * 3 * HID; Kw = 3 * HID; koff = (op & 1) * 32; }
  else if (op < 3 * ROUNDS) { src = eW2 + (op - 2 * ROUNDS) * HID * HID; Kw = HID; }
  else if (op < 4 * ROUNDS) { src = nW1 + (op - 3 * ROUNDS) * HID * 2 * HID; Kw = 2 * HID; }
  else if (op < 5 * ROUNDS) { src = nW2 + (op - 4 * ROUNDS) * HID * HID; Kw = HID; }
  else                      { src = dW1; Kw = HID; }
  int row = lane & 15, hi = lane >> 4;
  u16* dst = wsA + (size_t)op * 512 + lane * 16;
#pragma unroll
  for (int e = 0; e < 16; e++) {
    int kk = e + 16 * hi + koff;
    dst[e] = (kk < Kw) ? f2bf(src[row * Kw + kk]) : (u16)0;
  }
}

// ---------------- scalar->16 MLP encoder (1 -> h -> h), VALU ----------------
__global__ void encode_kernel(const float* __restrict__ x, int n,
                              const float* __restrict__ W1, const float* __restrict__ b1,
                              const float* __restrict__ W2, const float* __restrict__ b2,
                              u16* __restrict__ outf) {
  int i = blockIdx.x * blockDim.x + threadIdx.x;
  if (i >= n) return;
  float xv = x[i];
  float h[HID];
#pragma unroll
  for (int j = 0; j < HID; j++) h[j] = fmaxf(W1[j] * xv + b1[j], 0.f);
  u16* o = outf + (size_t)i * HID;
#pragma unroll
  for (int j = 0; j < HID; j++) {
    float acc = b2[j];
#pragma unroll
    for (int k = 0; k < HID; k++) acc += W2[j * HID + k] * h[k];
    o[j] = f2bf(acc);
  }
}

__global__ void deg_kernel(const int* __restrict__ recv, int E, float* __restrict__ deg) {
  int i = blockIdx.x * blockDim.x + threadIdx.x;
  if (i < E) atomicAdd(&deg[recv[i]], 1.0f);
}
__global__ void invdeg_kernel(float* __restrict__ deg, int N) {
  int i = blockIdx.x * blockDim.x + threadIdx.x;
  if (i < N) deg[i] = 1.0f / fmaxf(deg[i], 1.0f);
}

// ---------------- edge update: one wave = 16 edges ----------------
__global__ __launch_bounds__(128) void edge_update(
    const u16* __restrict__ nf, u16* __restrict__ ef, float* __restrict__ agg,
    const int* __restrict__ sends, const int* __restrict__ recvs,
    const u16* __restrict__ wsA, const float* __restrict__ b1,
    const float* __restrict__ b2, int E, int r) {
  int wave = threadIdx.x >> 5, lane = threadIdx.x & 31;
  int ebase = (blockIdx.x * 4 + wave) * 16;
  if (ebase >= E) return;
  int col = lane & 15, hi = lane >> 4;
  int e = ebase + col;
  int rv = recvs[e];
  int node = hi ? rv : sends[e];
  // B chunk0: k 0..15 = nf[sender] (lanes 0-15), k 16..31 = nf[receiver] (lanes 16-31)
  v16u bv0 = *(const v16u*)(nf + (size_t)node * HID);
  // B chunk1: k 32..47 = ef[e] (lanes 0-15), k 48..63 = zero pad (lanes 16-31)
  v16u bv1 = *(const v16u*)(ef + (size_t)e * HID);
  if (hi) {
#pragma unroll
    for (int i = 0; i < 16; i++) bv1[i] = (u16)0;
  }
  v16u a0 = *(const v16u*)(wsA + (size_t)(2 * r) * 512 + lane * 16);
  v16u a1 = *(const v16u*)(wsA + (size_t)(2 * r + 1) * 512 + lane * 16);
  const float* bb1 = b1 + r * HID + 8 * hi;
  v8f c;
#pragma unroll
  for (int i = 0; i < 8; i++) c[i] = bb1[i];
  c = wmma_bf16(a0, bv0, c);
  c = wmma_bf16(a1, bv1, c);
  // relu + repack H (C/D layout) into B layout for layer 2 (K=16, hi lanes padded)
  v16u hv;
#pragma unroll
  for (int i = 0; i < 8; i++) {
    float h = fmaxf(c[i], 0.f);
    float o = swapx16(h); // partner half-rows
    hv[i]     = hi ? (u16)0 : f2bf(h);
    hv[i + 8] = hi ? (u16)0 : f2bf(o);
  }
  const float* bb2 = b2 + r * HID + 8 * hi;
  v8f c2;
#pragma unroll
  for (int i = 0; i < 8; i++) c2[i] = bb2[i];
  v16u a2 = *(const v16u*)(wsA + (size_t)(10 + r) * 512 + lane * 16);
  c2 = wmma_bf16(a2, hv, c2);
  // store new ef (lane covers feats 8*hi..8*hi+7 of edge e) and scatter to agg
  v8u ov;
#pragma unroll
  for (int i = 0; i < 8; i++) ov[i] = f2bf(c2[i]);
  *(v8u*)(ef + (size_t)e * HID + 8 * hi) = ov;
  float* ag = agg + (size_t)rv * HID + 8 * hi;
#pragma unroll
  for (int i = 0; i < 8; i++) atomicAdd(ag + i, c2[i]);
}

// ---------------- node update: one wave = 16 nodes ----------------
__global__ __launch_bounds__(64) void node_update(
    u16* __restrict__ nf, float* __restrict__ agg, const float* __restrict__ invdeg,
    const u16* __restrict__ wsA, const float* __restrict__ b1,
    const float* __restrict__ b2, int N, int r) {
  int wave = threadIdx.x >> 5, lane = threadIdx.x & 31;
  int nbase = (blockIdx.x * 2 + wave) * 16;
  if (nbase >= N) return;
  int col = lane & 15, hi = lane >> 4;
  int n = nbase + col;
  v16u bv;
  if (!hi) { // k 0..15 = nf[n]
    bv = *(const v16u*)(nf + (size_t)n * HID);
  } else {   // k 16..31 = agg[n] / deg[n]
    float inv = invdeg[n];
    const float* ag = agg + (size_t)n * HID;
#pragma unroll
    for (int i = 0; i < 16; i++) bv[i] = f2bf(ag[i] * inv);
  }
  const float* bb1 = b1 + r * HID + 8 * hi;
  v8f c;
#pragma unroll
  for (int i = 0; i < 8; i++) c[i] = bb1[i];
  v16u a0 = *(const v16u*)(wsA + (size_t)(15 + r) * 512 + lane * 16);
  c = wmma_bf16(a0, bv, c);
  v16u hv;
#pragma unroll
  for (int i = 0; i < 8; i++) {
    float h = fmaxf(c[i], 0.f);
    float o = swapx16(h);
    hv[i]     = hi ? (u16)0 : f2bf(h);
    hv[i + 8] = hi ? (u16)0 : f2bf(o);
  }
  const float* bb2 = b2 + r * HID + 8 * hi;
  v8f c2;
#pragma unroll
  for (int i = 0; i < 8; i++) c2[i] = bb2[i];
  v16u a1 = *(const v16u*)(wsA + (size_t)(20 + r) * 512 + lane * 16);
  c2 = wmma_bf16(a1, hv, c2);
  v8u ov;
#pragma unroll
  for (int i = 0; i < 8; i++) ov[i] = f2bf(c2[i]);
  *(v8u*)(nf + (size_t)n * HID + 8 * hi) = ov;
  if (hi) { // reset agg row for next round (this lane owns node n)
    float* ag = agg + (size_t)n * HID;
#pragma unroll
    for (int i = 0; i < 16; i++) ag[i] = 0.f;
  }
}

// ---------------- bidirectional edge averaging ----------------
__global__ void bi_avg(const int* __restrict__ bi, int nbi, u16* __restrict__ ef) {
  int k = blockIdx.x * blockDim.x + threadIdx.x;
  if (k >= nbi) return;
  int i = bi[k], j = bi[nbi + k];
  u16* a = ef + (size_t)i * HID;
  u16* b = ef + (size_t)j * HID;
#pragma unroll
  for (int t = 0; t < HID; t++) {
    u16 v = f2bf(0.5f * (bf2f(a[t]) + bf2f(b[t])));
    a[t] = v; b[t] = v;
  }
}

// ---------------- decoder: h -> h -> 1 with tril mask ----------------
__global__ __launch_bounds__(128) void decode_kernel(
    const u16* __restrict__ ef, const int* __restrict__ recvs, const int* __restrict__ sends,
    const u16* __restrict__ wsA, const float* __restrict__ b1,
    const float* __restrict__ W2, const float* __restrict__ b2,
    float* __restrict__ out, int E) {
  int wave = threadIdx.x >> 5, lane = threadIdx.x & 31;
  int ebase = (blockIdx.x * 4 + wave) * 16;
  if (ebase >= E) return;
  int col = lane & 15, hi = lane >> 4;
  int e = ebase + col;
  v16u bv = *(const v16u*)(ef + (size_t)e * HID);
  if (hi) {
#pragma unroll
    for (int i = 0; i < 16; i++) bv[i] = (u16)0;
  }
  const float* bb1 = b1 + 8 * hi;
  v8f c;
#pragma unroll
  for (int i = 0; i < 8; i++) c[i] = bb1[i];
  v16u a0 = *(const v16u*)(wsA + (size_t)25 * 512 + lane * 16);
  c = wmma_bf16(a0, bv, c);
  // final dot: val[col] = sum_m W2[m] * relu(H[m][col]) + b2
  float p = 0.f;
#pragma unroll
  for (int i = 0; i < 8; i++) p += W2[8 * hi + i] * fmaxf(c[i], 0.f);
  p += swapx16(p); // combine the two half-row partial sums
  if (!hi) {
    float val = p + b2[0];
    out[e] = (recvs[e] >= sends[e]) ? val : 0.f;
  }
}

extern "C" void kernel_launch(void* const* d_in, const int* in_sizes, int n_in,
                              void* d_out, int out_size, void* d_ws, size_t ws_size,
                              hipStream_t stream) {
  (void)n_in; (void)out_size; (void)ws_size;
  const float* nodes = (const float*)d_in[0];
  const float* edges = (const float*)d_in[1];
  const int*   recvs = (const int*)d_in[2];
  const int*   sends = (const int*)d_in[3];
  const int*   bi    = (const int*)d_in[4];
  const float* neW1 = (const float*)d_in[5],  *neb1 = (const float*)d_in[6];
  const float* neW2 = (const float*)d_in[7],  *neb2 = (const float*)d_in[8];
  const float* eeW1 = (const float*)d_in[9],  *eeb1 = (const float*)d_in[10];
  const float* eeW2 = (const float*)d_in[11], *eeb2 = (const float*)d_in[12];
  const float* meW1 = (const float*)d_in[13], *meb1 = (const float*)d_in[14];
  const float* meW2 = (const float*)d_in[15], *meb2 = (const float*)d_in[16];
  const float* mnW1 = (const float*)d_in[17], *mnb1 = (const float*)d_in[18];
  const float* mnW2 = (const float*)d_in[19], *mnb2 = (const float*)d_in[20];
  const float* dW1  = (const float*)d_in[21], *db1  = (const float*)d_in[22];
  const float* dW2  = (const float*)d_in[23], *db2  = (const float*)d_in[24];

  const int N = in_sizes[0];
  const int E = in_sizes[1];
  const int NBI = in_sizes[4] / 2;

  // carve workspace (256B aligned slabs): weights | nf bf16 | ef bf16 | agg f32 | deg f32
  char* p = (char*)d_ws;
  auto carve = [&](size_t bytes) -> char* {
    char* q = p;
    p += (bytes + 255) & ~(size_t)255;
    return q;
  };
  u16*   wsA    = (u16*)carve((size_t)NOPS * 512 * sizeof(u16));
  u16*   nf     = (u16*)carve((size_t)N * HID * sizeof(u16));
  u16*   ef     = (u16*)carve((size_t)E * HID * sizeof(u16));
  float* agg    = (float*)carve((size_t)N * HID * sizeof(float));
  float* invdeg = (float*)carve((size_t)N * sizeof(float));

  hipMemsetAsync(agg, 0, (size_t)N * HID * sizeof(float), stream);
  hipMemsetAsync(invdeg, 0, (size_t)N * sizeof(float), stream);

  prep_weights<<<NOPS, 32, 0, stream>>>(meW1, meW2, mnW1, mnW2, dW1, wsA);

  encode_kernel<<<(N + 255) / 256, 256, 0, stream>>>(nodes, N, neW1, neb1, neW2, neb2, nf);
  encode_kernel<<<(E + 255) / 256, 256, 0, stream>>>(edges, E, eeW1, eeb1, eeW2, eeb2, ef);

  deg_kernel<<<(E + 255) / 256, 256, 0, stream>>>(recvs, E, invdeg);
  invdeg_kernel<<<(N + 255) / 256, 256, 0, stream>>>(invdeg, N);

  int nwE = (E + 15) / 16, nwN = (N + 15) / 16;
  for (int r = 0; r < ROUNDS; r++) {
    edge_update<<<(nwE + 3) / 4, 128, 0, stream>>>(nf, ef, agg, sends, recvs, wsA,
                                                   meb1, meb2, E, r);
    node_update<<<(nwN + 1) / 2, 64, 0, stream>>>(nf, agg, invdeg, wsA, mnb1, mnb2, N, r);
  }

  bi_avg<<<(NBI + 255) / 256, 256, 0, stream>>>(bi, NBI, ef);

  decode_kernel<<<(nwE + 3) / 4, 128, 0, stream>>>(ef, recvs, sends, wsA, db1, dW2, db2,
                                                   (float*)d_out, E);

  // outputs 2 & 3: receivers, senders (raw int32 bits into the output buffer)
  hipMemcpyAsync((char*)d_out + (size_t)E * 4, (const void*)recvs, (size_t)E * 4,
                 hipMemcpyDeviceToDevice, stream);
  hipMemcpyAsync((char*)d_out + (size_t)E * 8, (const void*)sends, (size_t)E * 4,
                 hipMemcpyDeviceToDevice, stream);
}
